// HeteroGNNBaseline_46901042872931
// MI455X (gfx1250) — matrix operand order repaired and checked by reading, not compiled, not run on hardware
//
#include <hip/hip_runtime.h>
#include <hip/hip_bf16.h>

// ---------------------------------------------------------------------------
// HeteroGNN (2-layer SAGE, 3 relations) for MI455X / gfx1250.
//  - scatter-mean via global f32 atomics (bandwidth-bound part, ~2.5 GB)
//  - all dense math via v_wmma_f32_16x16x32_bf16 (f32 -> bf16 via native
//    fptrunc/v_cvt_pk_bf16_f32; weights pre-swizzled to B-fragment layout)
// ---------------------------------------------------------------------------

#define N_NODES 100000
#define N_EDGES 400000
#define N_REL   3
#define DIM     128
// 16x16 output tiles: 8 col-tiles (ct), 4 k-tiles (kt) per 128x128 matrix
#define FRAG_ELEMS (8 * 4 * 32 * 16)   // 16384 bf16 per swizzled 128x128 matrix

typedef __attribute__((ext_vector_type(16))) __bf16          v16bf;
typedef __attribute__((ext_vector_type(16))) unsigned short  v16u16;
typedef __attribute__((ext_vector_type(8)))  float           v8f;

// Native RNE conversion: clang lowers fptrunc f32->bf16 to the hardware
// packed/scalar bf16 convert on gfx1250 (much cheaper than bit-twiddled RNE).
__device__ __forceinline__ __bf16 f2bf(float f) { return (__bf16)f; }

__device__ __forceinline__ unsigned short f2bf_bits(float f) {
    return __builtin_bit_cast(unsigned short, (__bf16)f);
}

// A fragment (16x32 bf16): base_row points at start of this lane's row.
// lane<16: K = kt*32 + {0..7, 16..23}; lane>=16: +8 shift (khalf).
__device__ __forceinline__ v16bf load_a_frag(const float* __restrict__ base_row,
                                             int kt, int khalf, float s) {
    const float* p = base_row + kt * 32 + khalf * 8;
    const float4 x0 = *(const float4*)(p + 0);
    const float4 x1 = *(const float4*)(p + 4);
    const float4 x2 = *(const float4*)(p + 16);
    const float4 x3 = *(const float4*)(p + 20);
    v16bf a;
    a[0]  = f2bf(x0.x * s); a[1]  = f2bf(x0.y * s);
    a[2]  = f2bf(x0.z * s); a[3]  = f2bf(x0.w * s);
    a[4]  = f2bf(x1.x * s); a[5]  = f2bf(x1.y * s);
    a[6]  = f2bf(x1.z * s); a[7]  = f2bf(x1.w * s);
    a[8]  = f2bf(x2.x * s); a[9]  = f2bf(x2.y * s);
    a[10] = f2bf(x2.z * s); a[11] = f2bf(x2.w * s);
    a[12] = f2bf(x3.x * s); a[13] = f2bf(x3.y * s);
    a[14] = f2bf(x3.z * s); a[15] = f2bf(x3.w * s);
    return a;
}

// B fragment: pre-swizzled [ct][kt][lane][16] -> one contiguous 32B load.
__device__ __forceinline__ v16bf load_b_frag(const unsigned short* __restrict__ W,
                                             int ct, int kt, int lane) {
    const v16u16* q = (const v16u16*)(W + ((((ct * 4 + kt) * 32) + lane) << 4));
    return __builtin_bit_cast(v16bf, *q);
}

// ---------------------------------------------------------------------------
// Weight prep: swizzle Wn[l][r] (dest layout [l][r][ct][kt][lane][i])
__global__ __launch_bounds__(256) void prep_wn_kernel(
    const float* __restrict__ Wn, unsigned short* __restrict__ dst) {
    int gid = blockIdx.x * blockDim.x + threadIdx.x;
    if (gid >= 2 * 3 * FRAG_ELEMS) return;
    int t = gid;
    int i    = t & 15;  t >>= 4;
    int lane = t & 31;  t >>= 5;
    int kt   = t & 3;   t >>= 2;
    int ct   = t & 7;   t >>= 3;
    int r    = t % 3;
    int l    = t / 3;
    int col  = ct * 16 + (lane & 15);
    int k    = kt * 32 + ((lane >> 4) << 3) + (i & 7) + ((i >> 3) << 4);
    float v  = Wn[(((size_t)(l * 3 + r) * DIM) + k) * DIM + col];
    dst[gid] = f2bf_bits(v);
}

// Wsum[l] = sum_r Wr[l][r], swizzled (dest layout [l][ct][kt][lane][i])
__global__ __launch_bounds__(256) void prep_wsum_kernel(
    const float* __restrict__ Wr, unsigned short* __restrict__ dst) {
    int gid = blockIdx.x * blockDim.x + threadIdx.x;
    if (gid >= 2 * FRAG_ELEMS) return;
    int t = gid;
    int i    = t & 15;  t >>= 4;
    int lane = t & 31;  t >>= 5;
    int kt   = t & 3;   t >>= 2;
    int ct   = t & 7;   t >>= 3;
    int l    = t;
    int col  = ct * 16 + (lane & 15);
    int k    = kt * 32 + ((lane >> 4) << 3) + (i & 7) + ((i >> 3) << 4);
    float v = 0.f;
    #pragma unroll
    for (int r = 0; r < 3; ++r)
        v += Wr[(((size_t)(l * 3 + r) * DIM) + k) * DIM + col];
    dst[gid] = f2bf_bits(v);
}

__global__ __launch_bounds__(256) void prep_bsum_kernel(
    const float* __restrict__ b, float* __restrict__ bsum) {
    int gid = blockIdx.x * blockDim.x + threadIdx.x;
    if (gid >= 2 * DIM) return;
    int l = gid >> 7, n = gid & 127;
    bsum[gid] = b[(l * 3 + 0) * DIM + n] + b[(l * 3 + 1) * DIM + n] +
                b[(l * 3 + 2) * DIM + n];
}

// ---------------------------------------------------------------------------
// Degree counts + reciprocal
__global__ __launch_bounds__(256) void count_kernel(
    const int* __restrict__ ei, float* __restrict__ cnt) {
    int gid = blockIdx.x * blockDim.x + threadIdx.x;
    if (gid >= N_REL * N_EDGES) return;
    int r = gid / N_EDGES, e = gid - r * N_EDGES;
    int dst = ei[((size_t)r * 2 + 1) * N_EDGES + e];
    atomicAdd(&cnt[(size_t)r * N_NODES + dst], 1.0f);
}

__global__ __launch_bounds__(256) void rcnt_kernel(
    const float* __restrict__ cnt, float* __restrict__ rcnt) {
    int gid = blockIdx.x * blockDim.x + threadIdx.x;
    if (gid >= N_REL * N_NODES) return;
    rcnt[gid] = 1.0f / fmaxf(cnt[gid], 1.0f);
}

// ---------------------------------------------------------------------------
// Scatter-add neighbor features: one wave (32 lanes) per edge, 4 dims/lane.
__global__ __launch_bounds__(256) void scatter_kernel(
    const int* __restrict__ ei, const float* __restrict__ h,
    float* __restrict__ agg) {
    long long gid = (long long)blockIdx.x * blockDim.x + threadIdx.x;
    long long eg  = gid >> 5;
    if (eg >= (long long)N_REL * N_EDGES) return;
    int lane = (int)(gid & 31);
    int r = (int)(eg / N_EDGES);
    int e = (int)(eg - (long long)r * N_EDGES);
    int src = ei[((size_t)r * 2 + 0) * N_EDGES + e];
    int dst = ei[((size_t)r * 2 + 1) * N_EDGES + e];
    float4 v = *((const float4*)(h + (size_t)src * DIM) + lane);
    float* ap = agg + ((size_t)r * N_NODES + dst) * DIM + lane * 4;
    atomicAdd(ap + 0, v.x);
    atomicAdd(ap + 1, v.y);
    atomicAdd(ap + 2, v.z);
    atomicAdd(ap + 3, v.w);
}

// ---------------------------------------------------------------------------
// Fused layer GEMM: out = relu?( h@Wsum + Sum_r (agg_r/cnt_r)@Wn_r + bsum )
// block = 128 threads = 4 waves; block tile 16 rows x 128 cols;
// wave tile 16x32 (2 accumulators); K loop 4 x 32 per matrix; 32 WMMA/wave.
__global__ __launch_bounds__(128) void layer_kernel(
    const float* __restrict__ h, const float* __restrict__ agg,
    const float* __restrict__ rcnt,
    const unsigned short* __restrict__ Wsum,   // swizzled [8][4][32][16]
    const unsigned short* __restrict__ Wn,     // swizzled [3][8][4][32][16]
    const float* __restrict__ bsum,            // [128]
    float* __restrict__ out, int do_relu) {
    const int lane  = threadIdx.x & 31;
    const int wave  = threadIdx.x >> 5;
    const int rowB  = blockIdx.x * 16;
    const int row   = rowB + (lane & 15);
    const int khalf = lane >> 4;
    const int ct0   = wave * 2;
    const int ct1   = ct0 + 1;

    v8f acc0 = {0.f, 0.f, 0.f, 0.f, 0.f, 0.f, 0.f, 0.f};
    v8f acc1 = {0.f, 0.f, 0.f, 0.f, 0.f, 0.f, 0.f, 0.f};

    // root term: h @ Wsum
    {
        const float* br = h + (size_t)row * DIM;
        #pragma unroll
        for (int kt = 0; kt < 4; ++kt) {
            v16bf a  = load_a_frag(br, kt, khalf, 1.0f);
            v16bf b0 = load_b_frag(Wsum, ct0, kt, lane);
            v16bf b1 = load_b_frag(Wsum, ct1, kt, lane);
            acc0 = __builtin_amdgcn_wmma_f32_16x16x32_bf16(false, a, false, b0,
                                                           (short)0, acc0, false, false);
            acc1 = __builtin_amdgcn_wmma_f32_16x16x32_bf16(false, a, false, b1,
                                                           (short)0, acc1, false, false);
        }
    }
    // neighbor terms: (agg_r * rcnt_r) @ Wn_r  (mean fused into A load)
    #pragma unroll
    for (int r = 0; r < N_REL; ++r) {
        const float* br = agg + ((size_t)r * N_NODES + row) * DIM;
        const float  s  = rcnt[(size_t)r * N_NODES + row];
        const unsigned short* W = Wn + (size_t)r * FRAG_ELEMS;
        #pragma unroll
        for (int kt = 0; kt < 4; ++kt) {
            v16bf a  = load_a_frag(br, kt, khalf, s);
            v16bf b0 = load_b_frag(W, ct0, kt, lane);
            v16bf b1 = load_b_frag(W, ct1, kt, lane);
            acc0 = __builtin_amdgcn_wmma_f32_16x16x32_bf16(false, a, false, b0,
                                                           (short)0, acc0, false, false);
            acc1 = __builtin_amdgcn_wmma_f32_16x16x32_bf16(false, a, false, b1,
                                                           (short)0, acc1, false, false);
        }
    }

    // epilogue: bias (+ReLU), scatter C/D layout back to row-major f32
    const int c0 = wave * 32 + (lane & 15);
    const int c1 = c0 + 16;
    const int r0 = rowB + ((lane >> 4) << 3);
    const float bb0 = bsum[c0];
    const float bb1 = bsum[c1];
    #pragma unroll
    for (int i = 0; i < 8; ++i) {
        float v0 = acc0[i] + bb0;
        float v1 = acc1[i] + bb1;
        if (do_relu) { v0 = fmaxf(v0, 0.f); v1 = fmaxf(v1, 0.f); }
        out[(size_t)(r0 + i) * DIM + c0] = v0;
        out[(size_t)(r0 + i) * DIM + c1] = v1;
    }
}

// ---------------------------------------------------------------------------
// Classifier: out[n, 0:2] = h @ Wc + bc   (trivial FLOPs, bandwidth pass)
__global__ __launch_bounds__(256) void classifier_kernel(
    const float* __restrict__ h, const float* __restrict__ Wc,
    const float* __restrict__ bc, float* __restrict__ out) {
    int n = blockIdx.x * blockDim.x + threadIdx.x;
    if (n >= N_NODES) return;
    const float4* hr = (const float4*)(h + (size_t)n * DIM);
    float a0 = bc[0], a1 = bc[1];
    #pragma unroll 8
    for (int d4 = 0; d4 < DIM / 4; ++d4) {
        float4 v = hr[d4];
        int d = d4 * 4;
        a0 += v.x * Wc[(d + 0) * 2 + 0] + v.y * Wc[(d + 1) * 2 + 0] +
              v.z * Wc[(d + 2) * 2 + 0] + v.w * Wc[(d + 3) * 2 + 0];
        a1 += v.x * Wc[(d + 0) * 2 + 1] + v.y * Wc[(d + 1) * 2 + 1] +
              v.z * Wc[(d + 2) * 2 + 1] + v.w * Wc[(d + 3) * 2 + 1];
    }
    out[(size_t)n * 2 + 0] = a0;
    out[(size_t)n * 2 + 1] = a1;
}

// ---------------------------------------------------------------------------
extern "C" void kernel_launch(void* const* d_in, const int* in_sizes, int n_in,
                              void* d_out, int out_size, void* d_ws, size_t ws_size,
                              hipStream_t stream) {
    const float* x   = (const float*)d_in[0];   // [N, 128]
    const int*   ei  = (const int*)  d_in[1];   // [3, 2, E]
    const float* Wn  = (const float*)d_in[2];   // [2, 3, 128, 128]
    const float* Wr  = (const float*)d_in[3];   // [2, 3, 128, 128]
    const float* b   = (const float*)d_in[4];   // [2, 3, 128]
    const float* Wc  = (const float*)d_in[5];   // [128, 2]
    const float* bc  = (const float*)d_in[6];   // [2]
    float* out = (float*)d_out;                 // [N, 2]

    // workspace carve-up (256B aligned)
    size_t off = 0;
    auto carve = [&](size_t bytes) {
        size_t o = off;
        off += (bytes + 255) & ~(size_t)255;
        return o;
    };
    char* ws = (char*)d_ws;
    float*          agg     = (float*)(ws + carve((size_t)N_REL * N_NODES * DIM * 4));
    float*          h1      = (float*)(ws + carve((size_t)N_NODES * DIM * 4));
    float*          h2      = (float*)(ws + carve((size_t)N_NODES * DIM * 4));
    float*          cnt     = (float*)(ws + carve((size_t)N_REL * N_NODES * 4));
    float*          rcnt    = (float*)(ws + carve((size_t)N_REL * N_NODES * 4));
    unsigned short* Wn_sw   = (unsigned short*)(ws + carve((size_t)2 * 3 * FRAG_ELEMS * 2));
    unsigned short* Wsum_sw = (unsigned short*)(ws + carve((size_t)2 * FRAG_ELEMS * 2));
    float*          bsum    = (float*)(ws + carve((size_t)2 * DIM * 4));
    (void)ws_size; (void)in_sizes; (void)n_in; (void)out_size;

    const int TB = 256;

    // degree counts (edge structure is layer-invariant)
    hipMemsetAsync(cnt, 0, (size_t)N_REL * N_NODES * 4, stream);
    count_kernel<<<(N_REL * N_EDGES + TB - 1) / TB, TB, 0, stream>>>(ei, cnt);
    rcnt_kernel<<<(N_REL * N_NODES + TB - 1) / TB, TB, 0, stream>>>(cnt, rcnt);

    // weight prep (tiny)
    prep_wn_kernel<<<(2 * 3 * FRAG_ELEMS + TB - 1) / TB, TB, 0, stream>>>(Wn, Wn_sw);
    prep_wsum_kernel<<<(2 * FRAG_ELEMS + TB - 1) / TB, TB, 0, stream>>>(Wr, Wsum_sw);
    prep_bsum_kernel<<<1, 256, 0, stream>>>(b, bsum);

    const long long scatter_threads = (long long)N_REL * N_EDGES * 32;
    const int scatter_blocks = (int)((scatter_threads + TB - 1) / TB);
    const int gemm_blocks = N_NODES / 16;   // 100000 = 6250 * 16 exactly

    // ---- layer 0: input x, ReLU ----
    hipMemsetAsync(agg, 0, (size_t)N_REL * N_NODES * DIM * 4, stream);
    scatter_kernel<<<scatter_blocks, TB, 0, stream>>>(ei, x, agg);
    layer_kernel<<<gemm_blocks, 128, 0, stream>>>(
        x, agg, rcnt, Wsum_sw + 0 * FRAG_ELEMS, Wn_sw + 0 * 3 * FRAG_ELEMS,
        bsum + 0 * DIM, h1, /*relu=*/1);

    // ---- layer 1: input h1, no activation ----
    hipMemsetAsync(agg, 0, (size_t)N_REL * N_NODES * DIM * 4, stream);
    scatter_kernel<<<scatter_blocks, TB, 0, stream>>>(ei, h1, agg);
    layer_kernel<<<gemm_blocks, 128, 0, stream>>>(
        h1, agg, rcnt, Wsum_sw + 1 * FRAG_ELEMS, Wn_sw + 1 * 3 * FRAG_ELEMS,
        bsum + 1 * DIM, h2, /*relu=*/0);

    // ---- classifier ----
    classifier_kernel<<<(N_NODES + TB - 1) / TB, TB, 0, stream>>>(h2, Wc, bc, out);
}
